// MaxSimilarityTTADualSoftMaxMatcher_61014305407066
// MI455X (gfx1250) — compile-verified, matrix operand order checked by reading, not run on hardware
//
#include <hip/hip_runtime.h>

// ---------------------------------------------------------------------------
// MaxSimilarityTTADualSoftMaxMatcher on MI455X (gfx1250), wave32 + WMMA bf16.
//
// Problem: B=2, NA=NB=4096, D=256, K=8 steerers -> 17 similarity variants.
//   corr = 10 * max_j ( A @ Baug_j^T ),  Baug_j in {B, B W_k^T, B W_k}
//   P = exp(2c + s_n + t_m);  mutual via log-space argmax equalities.
//
// k_corr: block = 8 waves = 128x32 output tile. Per variant the block stages
// the 32x256 bf16 B-tile (16 KB) into LDS with one coalesced b128 copy; all 8
// waves build WMMA operand fragments from LDS (8x less L2 B-traffic,
// AI ~ 106 flops/byte from L2 -> compute bound on v_wmma_f32_16x16x32_bf16).
//
// Workspace map (assumes ws_size >= ~203 MiB):
//   [0,   4MiB)  A bf16            [8192][256]
//   [4,  72MiB)  Baug bf16         [17][8192][256]   (slot 0 = B)
//   [72, 73MiB)  W  bf16           [8][256][256]     (row e -> W[e,d] for B@W^T)
//   [73, 74MiB)  W^T bf16          [8][256][256]     (row e -> W[d,e] for B@W)
//   [74,202MiB)  corr f32          [2][4096][4096]
//   [202MiB...)  s,t,cm,rbest,cbest (8192 f32 each) + 3x partial arrays [8][8192]
// ---------------------------------------------------------------------------

typedef __bf16 bf16t;
typedef bf16t v16bf __attribute__((ext_vector_type(16)));
typedef float v8f __attribute__((ext_vector_type(8)));

#define NPTS 4096
#define DDIM 256
#define NVAR 17
#define NM 33554432   /* 2*4096*4096 */
#define BPAD 264      /* 256 + 8 shorts: 528B row stride, keeps 16B align, rotates banks */

static __device__ __forceinline__ unsigned short f2bf(float f) {
  unsigned int u = __float_as_uint(f);
  u += 0x7fffu + ((u >> 16) & 1u);   // round to nearest even
  return (unsigned short)(u >> 16);
}

union FragU { uint4 q[2]; v16bf v; };

// Fragment load per CDNA5 ISA 7.12.2 (16-bit A/B 16x32 layout, wave32):
// lanes 0-15 hold K = {kb+0..7, kb+16..23}, lanes 16-31 hold K = {kb+8..15, kb+24..31}
static __device__ __forceinline__ v16bf load_frag(const unsigned short* mat, int row,
                                                  int kb, int lane) {
  int base0 = (lane >> 4) << 3;  // 0 or 8
  const unsigned short* p = mat + (size_t)row * DDIM + kb + base0;
  FragU f;
  f.q[0] = *(const uint4*)(p);        // 8 bf16: K = kb+base0 .. +7
  f.q[1] = *(const uint4*)(p + 16);   // 8 bf16: K = kb+16+base0 .. +7
  return f.v;
}

// ---------------------------------------------------------------- conversions
__global__ void __launch_bounds__(256) k_cvt(const float* __restrict__ src,
                                             unsigned short* __restrict__ dst, int n) {
  int i = blockIdx.x * 256 + threadIdx.x;
  if (i < n) dst[i] = f2bf(src[i]);
}

__global__ void __launch_bounds__(256) k_cvtW(const float* __restrict__ W,
                                              unsigned short* __restrict__ Wbf,
                                              unsigned short* __restrict__ Wt) {
  int i = blockIdx.x * 256 + threadIdx.x;          // < 8*256*256
  int k = i >> 16, rem = i & 65535, d = rem >> 8, e = rem & 255;
  unsigned short v = f2bf(W[i]);                   // W_k[d,e]
  Wbf[i] = v;                                      // row d of W (as stored)
  Wt[(k << 16) + (e << 8) + d] = v;                // transposed copy
}

// ---------------------------------------------- Baug = B @ {W^T, W}, 16 GEMMs
// One wave per 16x32 output tile. 65536 waves total (tiny vs corr GEMM).
__global__ void __launch_bounds__(256) k_baug(const unsigned short* __restrict__ Bbf,
                                              const unsigned short* __restrict__ Wbf,
                                              const unsigned short* __restrict__ Wt,
                                              unsigned short* __restrict__ Baug) {
  int lane = threadIdx.x & 31;
  int wid = (blockIdx.x << 3) + (threadIdx.x >> 5);  // 0..65535
  int v = wid >> 12;                                 // variant 0..15 -> slot v+1
  int rem = wid & 4095;
  int rt = rem >> 3;                                 // 0..511 row tiles
  int ct = rem & 7;                                  // 0..7 col tiles (32 wide)
  // odd slot (v even): B @ W_k^T -> operand rows are rows of W storage.
  // even slot (v odd): B @ W_k   -> operand rows are rows of W^T copy.
  const unsigned short* Y = (((v & 1) == 0) ? Wbf : Wt) + ((size_t)(v >> 1) << 16);
  int r0 = rt << 4, e0 = ct << 5;
  int arow = r0 + (lane & 15);
  int b0row = e0 + (lane & 15);
  v8f acc0 = {0.f, 0.f, 0.f, 0.f, 0.f, 0.f, 0.f, 0.f};
  v8f acc1 = acc0;
#pragma unroll
  for (int kc = 0; kc < 8; ++kc) {
    int kb = kc << 5;
    v16bf af = load_frag(Bbf, arow, kb, lane);
    v16bf bf0 = load_frag(Y, b0row, kb, lane);
    v16bf bf1 = load_frag(Y, b0row + 16, kb, lane);
    acc0 = __builtin_amdgcn_wmma_f32_16x16x32_bf16(false, af, false, bf0, (short)0, acc0, false, false);
    acc1 = __builtin_amdgcn_wmma_f32_16x16x32_bf16(false, af, false, bf1, (short)0, acc1, false, false);
  }
  unsigned short* out = Baug + ((size_t)(v + 1) << 21);  // slot (v+1)*8192*256
  int nn = lane & 15;
  int mhi = (lane >> 4) << 3;
#pragma unroll
  for (int r = 0; r < 8; ++r) {
    int row = r0 + r + mhi;
    out[(size_t)row * DDIM + e0 + nn] = f2bf(acc0[r]);
    out[(size_t)row * DDIM + e0 + 16 + nn] = f2bf(acc1[r]);
  }
}

// ------------------------- corr = 10 * max over 17 variants of A @ Baug_j^T
// Block (8 waves) computes a 128x32 tile; B tile staged in LDS per variant.
__global__ void __launch_bounds__(256) k_corr(const unsigned short* __restrict__ Abf,
                                              const unsigned short* __restrict__ Baug,
                                              float* __restrict__ corr) {
  __shared__ __align__(16) unsigned short Btile[32 * BPAD];  // 16.5 KB
  int tid = threadIdx.x;
  int lane = tid & 31;
  int wave = tid >> 5;                               // 0..7
  int bid = blockIdx.x;                              // 0..8191
  int b = bid >> 12;                                 // batch
  int rem = bid & 4095;
  int nb = rem >> 7;                                 // 0..31 (128-row blocks)
  int mt = rem & 127;                                // 0..127 (32-col tiles)
  int n0 = (nb << 7) + (wave << 4);
  int m0 = mt << 5;
  const unsigned short* Ab = Abf + ((size_t)b << 20);  // batch offset 4096*256
  int arow = n0 + (lane & 15);
  v16bf afr[8];                                      // hoisted A fragments
#pragma unroll
  for (int kc = 0; kc < 8; ++kc) afr[kc] = load_frag(Ab, arow, kc << 5, lane);
  v8f mx0, mx1;
#pragma unroll
  for (int i = 0; i < 8; ++i) { mx0[i] = -3.0e38f; mx1[i] = -3.0e38f; }
  int base0 = (lane >> 4) << 3;
  int brow = lane & 15;
  for (int v = 0; v < NVAR; ++v) {
    // rows m0..m0+31 are one contiguous 16 KB block of Baug -> coalesced copy
    const uint4* src = (const uint4*)(Baug + ((size_t)v << 21) + ((size_t)b << 20)
                                      + ((size_t)m0 << 8));
    __syncthreads();                                 // protect prior-iter reads
#pragma unroll
    for (int i = 0; i < 4; ++i) {
      int idx = tid + (i << 8);                      // 1024 uint4 total
      int row = idx >> 5, col = idx & 31;            // 32 uint4 per 256-short row
      *((uint4*)(Btile + row * BPAD) + col) = src[idx];
    }
    __syncthreads();
    v8f acc0 = {0.f, 0.f, 0.f, 0.f, 0.f, 0.f, 0.f, 0.f};
    v8f acc1 = acc0;
#pragma unroll
    for (int kc = 0; kc < 8; ++kc) {
      int kb = kc << 5;
      const unsigned short* p0 = Btile + brow * BPAD + kb + base0;
      const unsigned short* p1 = p0 + 16 * BPAD;
      FragU f0, f1;
      f0.q[0] = *(const uint4*)(p0);
      f0.q[1] = *(const uint4*)(p0 + 16);
      f1.q[0] = *(const uint4*)(p1);
      f1.q[1] = *(const uint4*)(p1 + 16);
      acc0 = __builtin_amdgcn_wmma_f32_16x16x32_bf16(false, afr[kc], false, f0.v, (short)0, acc0, false, false);
      acc1 = __builtin_amdgcn_wmma_f32_16x16x32_bf16(false, afr[kc], false, f1.v, (short)0, acc1, false, false);
    }
#pragma unroll
    for (int i = 0; i < 8; ++i) {
      mx0[i] = fmaxf(mx0[i], acc0[i]);
      mx1[i] = fmaxf(mx1[i], acc1[i]);
    }
  }
  float* cb = corr + ((size_t)b << 24);  // b*4096*4096
  int nn = lane & 15;
  int mhi = (lane >> 4) << 3;
#pragma unroll
  for (int r = 0; r < 8; ++r) {
    int n = n0 + r + mhi;
    cb[(size_t)n * NPTS + m0 + nn] = 10.0f * mx0[r];        // INV_TEMP
    cb[(size_t)n * NPTS + m0 + 16 + nn] = 10.0f * mx1[r];
  }
}

// ------------------------------------------------------- row softmax stats s_n
__global__ void __launch_bounds__(256) k_rowstats(const float* __restrict__ corr,
                                                  float* __restrict__ s) {
  __shared__ float red[256];
  int r = blockIdx.x;                 // flattened row 0..8191
  int tid = threadIdx.x;
  const float* row = corr + (size_t)r * NPTS;
  float vals[16];
  float lm = -3.0e38f;
#pragma unroll
  for (int i = 0; i < 16; ++i) { vals[i] = row[tid + (i << 8)]; lm = fmaxf(lm, vals[i]); }
  red[tid] = lm; __syncthreads();
  for (int w = 128; w > 0; w >>= 1) { if (tid < w) red[tid] = fmaxf(red[tid], red[tid + w]); __syncthreads(); }
  float rm = red[0]; __syncthreads();
  float ls = 0.f;
#pragma unroll
  for (int i = 0; i < 16; ++i) ls += expf(vals[i] - rm);
  red[tid] = ls; __syncthreads();
  for (int w = 128; w > 0; w >>= 1) { if (tid < w) red[tid] += red[tid + w]; __syncthreads(); }
  if (tid == 0) s[r] = -rm - logf(red[0]);
}

// --------------------------------------- column stats (deterministic 2-stage)
__global__ void __launch_bounds__(256) k_colpartmax(const float* __restrict__ corr,
                                                    float* __restrict__ pm) {
  int tid = threadIdx.x;
  int cc = blockIdx.x & 15;   // column chunk (256 cols)
  int rc = blockIdx.x >> 4;   // row chunk (512 rows), 0..15, never crosses batch
  int m = (cc << 8) + tid;
  int r0 = rc << 9;
  const float* p = corr + (size_t)r0 * NPTS + m;
  float lm = -3.0e38f;
  for (int i = 0; i < 512; ++i) lm = fmaxf(lm, p[(size_t)i * NPTS]);
  pm[((size_t)(rc & 7) << 13) + (((size_t)(rc >> 3)) << 12) + m] = lm;
}

__global__ void __launch_bounds__(256) k_pfinmax(const float* __restrict__ pm,
                                                 float* __restrict__ dst) {
  int bm = blockIdx.x * 256 + threadIdx.x;  // < 8192
  float v = pm[bm];
#pragma unroll
  for (int p = 1; p < 8; ++p) v = fmaxf(v, pm[((size_t)p << 13) + bm]);
  dst[bm] = v;
}

__global__ void __launch_bounds__(256) k_colpartsum(const float* __restrict__ corr,
                                                    const float* __restrict__ cm,
                                                    float* __restrict__ ps) {
  int tid = threadIdx.x;
  int cc = blockIdx.x & 15;
  int rc = blockIdx.x >> 4;
  int m = (cc << 8) + tid;
  int r0 = rc << 9;
  int bm = ((rc >> 3) << 12) + m;
  float cmax = cm[bm];
  const float* p = corr + (size_t)r0 * NPTS + m;
  float acc = 0.f;
  for (int i = 0; i < 512; ++i) acc += expf(p[(size_t)i * NPTS] - cmax);
  ps[((size_t)(rc & 7) << 13) + bm] = acc;
}

__global__ void __launch_bounds__(256) k_colfinsum(const float* __restrict__ ps,
                                                   const float* __restrict__ cm,
                                                   float* __restrict__ t) {
  int bm = blockIdx.x * 256 + threadIdx.x;
  float v = 0.f;
#pragma unroll
  for (int p = 0; p < 8; ++p) v += ps[((size_t)p << 13) + bm];
  t[bm] = -cm[bm] - logf(v);
}

// -------------------------------- log-space argmax stats for the mutual mask
__global__ void __launch_bounds__(256) k_rbest(const float* __restrict__ corr,
                                               const float* __restrict__ t,
                                               float* __restrict__ rbest) {
  __shared__ float red[256];
  int r = blockIdx.x;
  int tid = threadIdx.x;
  int bb = (r >> 12) << 12;   // 0 or 4096
  const float* row = corr + (size_t)r * NPTS;
  float lm = -3.0e38f;
#pragma unroll
  for (int i = 0; i < 16; ++i) {
    int m = tid + (i << 8);
    float v = 2.0f * row[m] + t[bb + m];   // 2c exact -> identical to k_final
    lm = fmaxf(lm, v);
  }
  red[tid] = lm; __syncthreads();
  for (int w = 128; w > 0; w >>= 1) { if (tid < w) red[tid] = fmaxf(red[tid], red[tid + w]); __syncthreads(); }
  if (tid == 0) rbest[r] = red[0];
}

__global__ void __launch_bounds__(256) k_cbestpart(const float* __restrict__ corr,
                                                   const float* __restrict__ s,
                                                   float* __restrict__ pb) {
  int tid = threadIdx.x;
  int cc = blockIdx.x & 15;
  int rc = blockIdx.x >> 4;
  int m = (cc << 8) + tid;
  int r0 = rc << 9;
  const float* p = corr + (size_t)r0 * NPTS + m;
  float lm = -3.0e38f;
  for (int i = 0; i < 512; ++i) {
    float v = 2.0f * p[(size_t)i * NPTS] + s[r0 + i];
    lm = fmaxf(lm, v);
  }
  pb[((size_t)(rc & 7) << 13) + (((size_t)(rc >> 3)) << 12) + m] = lm;
}

// ------------------------------------------------ final: P and mutual mask
__global__ void __launch_bounds__(256) k_final(const float* __restrict__ corr,
                                               const float* __restrict__ s,
                                               const float* __restrict__ t,
                                               const float* __restrict__ rbest,
                                               const float* __restrict__ cbest,
                                               float* __restrict__ out) {
  size_t idx = ((size_t)blockIdx.x << 8) + threadIdx.x;  // < 33,554,432
  int r = (int)(idx >> 12);
  int m = (int)(idx & 4095);
  int bm = ((r >> 12) << 12) + m;
  float c = corr[idx];
  float tm = t[bm], sn = s[r];
  float vT = 2.0f * c + tm;     // matches k_rbest expression bit-exactly
  float vS = 2.0f * c + sn;     // matches k_cbestpart expression bit-exactly
  float P = expf(vS + tm);      // exp(2c + s_n + t_m) = dual-softmax product
  bool mut = (vT == rbest[r]) && (vS == cbest[bm]) && (P > 0.0f);
  out[idx] = P;
  out[(size_t)NM + idx] = mut ? 1.0f : 0.0f;
}

// ---------------------------------------------------------------------------
extern "C" void kernel_launch(void* const* d_in, const int* in_sizes, int n_in,
                              void* d_out, int out_size, void* d_ws, size_t ws_size,
                              hipStream_t stream) {
  (void)in_sizes; (void)n_in; (void)out_size; (void)ws_size;
  const float* dA = (const float*)d_in[1];   // descriptions_A [2,4096,256]
  const float* dB = (const float*)d_in[3];   // descriptions_B [2,4096,256]
  const float* Wm = (const float*)d_in[4];   // steer_mats     [8,256,256]
  float* out = (float*)d_out;                // P then mutual, each 2*4096*4096

  char* ws = (char*)d_ws;
  const size_t MiB = (size_t)1 << 20;
  unsigned short* Abf  = (unsigned short*)(ws);
  unsigned short* Baug = (unsigned short*)(ws + 4 * MiB);
  unsigned short* Wbf  = (unsigned short*)(ws + 72 * MiB);
  unsigned short* Wt   = (unsigned short*)(ws + 73 * MiB);
  float* corr  = (float*)(ws + 74 * MiB);
  float* s     = (float*)(ws + 202 * MiB);
  float* t     = s + 8192;
  float* cm    = t + 8192;
  float* rbest = cm + 8192;
  float* cbest = rbest + 8192;
  float* pm    = cbest + 8192;   // [8][8192]
  float* ps    = pm + 65536;     // [8][8192]
  float* pb    = ps + 65536;     // [8][8192]

  k_cvt<<<8192, 256, 0, stream>>>(dA, Abf, 2097152);
  k_cvt<<<8192, 256, 0, stream>>>(dB, Baug, 2097152);       // Baug slot 0 = B
  k_cvtW<<<2048, 256, 0, stream>>>(Wm, Wbf, Wt);
  k_baug<<<8192, 256, 0, stream>>>(Baug, Wbf, Wt, Baug);    // slots 1..16
  k_corr<<<8192, 256, 0, stream>>>(Abf, Baug, corr);
  k_rowstats<<<8192, 256, 0, stream>>>(corr, s);
  k_colpartmax<<<256, 256, 0, stream>>>(corr, pm);
  k_pfinmax<<<32, 256, 0, stream>>>(pm, cm);
  k_colpartsum<<<256, 256, 0, stream>>>(corr, cm, ps);
  k_colfinsum<<<32, 256, 0, stream>>>(ps, cm, t);
  k_rbest<<<8192, 256, 0, stream>>>(corr, t, rbest);
  k_cbestpart<<<256, 256, 0, stream>>>(corr, s, pb);
  k_pfinmax<<<32, 256, 0, stream>>>(pb, cbest);
  k_final<<<131072, 256, 0, stream>>>(corr, s, t, rbest, cbest, out);
}
